// QuestionEmbedding_82738249990737
// MI455X (gfx1250) — matrix operand order repaired and checked by reading, not compiled, not run on hardware
//
#include <hip/hip_runtime.h>
#include <hip/hip_bf16.h>

typedef __attribute__((ext_vector_type(16))) __bf16 v16bf;
typedef __attribute__((ext_vector_type(8)))  __bf16 v8bf;
typedef __attribute__((ext_vector_type(8)))  float  v8f;

#define VOCAB 50000
#define EMB   512
#define HID   1024
#define BATCH 256
#define SEQW  26

union AFrag { v16bf v; v8bf h[2]; };

// ---------------- workspace layout (bytes) ----------------
// X  : bf16 [26][256][512]  = 6,815,744
// Wih: bf16 [4096][512]     = 4,194,304
// Whh: bf16 [4096][1024]    = 8,388,608
// hA : bf16 [256][1024]     =   524,288
// hB : bf16 [256][1024]     =   524,288
// C  : f32  [256][1024]     = 1,048,576
static const size_t OFF_X   = 0;
static const size_t OFF_WIH = 6815744;
static const size_t OFF_WHH = OFF_WIH + 4194304;
static const size_t OFF_HA  = OFF_WHH + 8388608;
static const size_t OFF_HB  = OFF_HA + 524288;
static const size_t OFF_C   = OFF_HB + 524288;

__global__ void cvt_f32_to_bf16(const float* __restrict__ src,
                                __bf16* __restrict__ dst, int n) {
    int i = blockIdx.x * blockDim.x + threadIdx.x;
    if (i < n) dst[i] = (__bf16)src[i];
}

// X[t][b][e] = tok>0 ? tanh(Wemb[e][tok-1]) : 0   (time-major, matches scan)
__global__ void embed_tanh_kernel(const int* __restrict__ ques,
                                  const float* __restrict__ Wemb,
                                  __bf16* __restrict__ X) {
    int idx = blockIdx.x * blockDim.x + threadIdx.x;   // exact grid: 26*256*512
    int e  = idx & (EMB - 1);
    int bt = idx >> 9;
    int b  = bt & (BATCH - 1);
    int t  = bt >> 8;
    int tok = ques[b * SEQW + t];
    float v = 0.0f;
    if (tok > 0) v = tanhf(Wemb[(size_t)e * VOCAB + (tok - 1)]);
    X[((size_t)t * BATCH + b) * EMB + e] = (__bf16)v;
}

struct Srcs { const __bf16* a; const __bf16* b; size_t gs; };

// per-iteration source pointers: iters 0..15 -> X/Wih panel, 16..47 -> H/Whh panel
__device__ __forceinline__ Srcs srcs_for(int it,
                                         const __bf16* aX, const __bf16* bX,
                                         const __bf16* aH, const __bf16* bH) {
    if (it < EMB / 32) {
        int k = it * 32;
        Srcs s; s.a = aX + k; s.b = bX + k; s.gs = (size_t)HID * EMB; return s;
    } else {
        int k = (it - EMB / 32) * 32;
        Srcs s; s.a = aH + k; s.b = bH + k; s.gs = (size_t)HID * HID; return s;
    }
}

__device__ __forceinline__ void load_frags(const Srcs& s, AFrag& a, v16bf b[4]) {
    a.h[0] = *(const v8bf*)(s.a);
    a.h[1] = *(const v8bf*)(s.a + 8);
    b[0] = *(const v16bf*)(s.b);
    b[1] = *(const v16bf*)(s.b + s.gs);
    b[2] = *(const v16bf*)(s.b + 2 * s.gs);
    b[3] = *(const v16bf*)(s.b + 3 * s.gs);
}

#define WMMA4(A, B)                                                                                   \
    do {                                                                                              \
        acc0 = __builtin_amdgcn_wmma_f32_16x16x32_bf16(false, (A).v, false, (B)[0], (short)0, acc0,   \
                                                       false, false);                                 \
        acc1 = __builtin_amdgcn_wmma_f32_16x16x32_bf16(false, (A).v, false, (B)[1], (short)0, acc1,   \
                                                       false, false);                                 \
        acc2 = __builtin_amdgcn_wmma_f32_16x16x32_bf16(false, (A).v, false, (B)[2], (short)0, acc2,   \
                                                       false, false);                                 \
        acc3 = __builtin_amdgcn_wmma_f32_16x16x32_bf16(false, (A).v, false, (B)[3], (short)0, acc3,   \
                                                       false, false);                                 \
    } while (0)

// One wave per workgroup: 16(batch) x 16(hid) tile, all four gates fused.
__global__ __launch_bounds__(32) __attribute__((amdgpu_waves_per_eu(1)))
void lstm_step_kernel(
        const __bf16* __restrict__ Xt,     // [256, 512]   bf16
        const __bf16* __restrict__ Hprev,  // [256, 1024]  bf16
        const __bf16* __restrict__ Wih,    // [4096, 512]  bf16
        const __bf16* __restrict__ Whh,    // [4096, 1024] bf16
        const float*  __restrict__ b_ih,   // [4096]
        const float*  __restrict__ b_hh,   // [4096]
        float*        __restrict__ C,      // [256, 1024]  f32, in-place
        __bf16*       __restrict__ Hnext,  // [256, 1024]  bf16
        float*        __restrict__ Hout,   // [256, 1024]  f32 (d_out)
        int write_out)
{
    const int lane = threadIdx.x & 31;
    const int l15  = lane & 15;
    const int lhi  = lane >> 4;            // 0 or 1

    const int mrow = blockIdx.y << 4;      // 16 batch rows   (grid.y = 16)
    const int ncol = blockIdx.x << 4;      // 16 hid cols     (grid.x = 64)

    const int aoff = lhi << 4;             // A K-offset per half-wave (contiguous, matches B)
    const int boff = lhi << 4;             // B K-offset per half-wave

    const __bf16* aX = Xt    + (size_t)(mrow + l15) * EMB + aoff;
    const __bf16* aH = Hprev + (size_t)(mrow + l15) * HID + aoff;
    const __bf16* bX = Wih   + (size_t)(ncol + l15) * EMB + boff;
    const __bf16* bH = Whh   + (size_t)(ncol + l15) * HID + boff;

    v8f acc0 = {}, acc1 = {}, acc2 = {}, acc3 = {};

    const int NITER = (EMB + HID) / 32;    // 48

    // software-pipelined K loop: iter k+1 fragments in flight under iter k WMMAs
    AFrag aC, aN;
    v16bf bC[4], bN[4];
    load_frags(srcs_for(0, aX, bX, aH, bH), aC, bC);

    #pragma unroll
    for (int it = 1; it < NITER; ++it) {
        load_frags(srcs_for(it, aX, bX, aH, bH), aN, bN);
        WMMA4(aC, bC);
        aC = aN;
        bC[0] = bN[0]; bC[1] = bN[1]; bC[2] = bN[2]; bC[3] = bN[3];
    }
    WMMA4(aC, bC);

#if __has_builtin(__builtin_amdgcn_sched_group_barrier)
    // Pin the pipeline: L(0), then { L(it), W(it-1) } for it=1..47, then W(47).
    // Each L = 10 global_load_b128 (2 A + 8 B), each W = 4 WMMAs.
    __builtin_amdgcn_sched_group_barrier(0x020, 10, 0);           // VMEM read: L(0)
    #pragma unroll
    for (int it = 1; it < NITER; ++it) {
        __builtin_amdgcn_sched_group_barrier(0x020, 10, 0);       // VMEM read: L(it)
        __builtin_amdgcn_sched_group_barrier(0x008, 4, 0);        // WMMA: W(it-1)
    }
    __builtin_amdgcn_sched_group_barrier(0x008, 4, 0);            // WMMA: W(47)
#endif

    // -------- fused LSTM cell update --------
    const int n = ncol + l15;
    const float bi = b_ih[n]           + b_hh[n];
    const float bf = b_ih[HID + n]     + b_hh[HID + n];
    const float bg = b_ih[2 * HID + n] + b_hh[2 * HID + n];
    const float bo = b_ih[3 * HID + n] + b_hh[3 * HID + n];

    #pragma unroll
    for (int v = 0; v < 8; ++v) {
        const int m = mrow + v + (lhi << 3);       // C/D layout: M = vgpr + 8*(lane>=16)
        const size_t idx = (size_t)m * HID + n;
        float gi = acc0[v] + bi;
        float gf = acc1[v] + bf;
        float gg = acc2[v] + bg;
        float go = acc3[v] + bo;
        float si = 1.0f / (1.0f + __expf(-gi));
        float sf = 1.0f / (1.0f + __expf(-gf));
        float so = 1.0f / (1.0f + __expf(-go));
        float cc = sf * C[idx] + si * tanhf(gg);
        C[idx] = cc;
        float hh = so * tanhf(cc);
        Hnext[idx] = (__bf16)hh;
        if (write_out) Hout[idx] = hh;
    }
}

extern "C" void kernel_launch(void* const* d_in, const int* in_sizes, int n_in,
                              void* d_out, int out_size, void* d_ws, size_t ws_size,
                              hipStream_t stream) {
    const int*   ques = (const int*)d_in[0];
    // d_in[1] = ques_len (unused by reference output)
    const float* Wemb = (const float*)d_in[2];
    const float* Wih  = (const float*)d_in[3];
    const float* Whh  = (const float*)d_in[4];
    const float* bih  = (const float*)d_in[5];
    const float* bhh  = (const float*)d_in[6];

    char* ws = (char*)d_ws;
    __bf16* X      = (__bf16*)(ws + OFF_X);
    __bf16* Wih_bf = (__bf16*)(ws + OFF_WIH);
    __bf16* Whh_bf = (__bf16*)(ws + OFF_WHH);
    __bf16* hA     = (__bf16*)(ws + OFF_HA);
    __bf16* hB     = (__bf16*)(ws + OFF_HB);
    float*  C      = (float*)(ws + OFF_C);

    // one-time (per call) parallel prep — no recurrence
    cvt_f32_to_bf16<<<(4 * HID * EMB) / 256, 256, 0, stream>>>(Wih, Wih_bf, 4 * HID * EMB);
    cvt_f32_to_bf16<<<(4 * HID * HID) / 256, 256, 0, stream>>>(Whh, Whh_bf, 4 * HID * HID);
    embed_tanh_kernel<<<(SEQW * BATCH * EMB) / 256, 256, 0, stream>>>(ques, Wemb, X);
    hipMemsetAsync(hA, 0, (size_t)BATCH * HID * sizeof(__bf16), stream);
    hipMemsetAsync(C,  0, (size_t)BATCH * HID * sizeof(float), stream);

    // 26 serialized fused GEMM+cell steps (h ping-pongs, c in place)
    // grid: 64 hid-tiles x 16 batch-tiles = 1024 single-wave workgroups
    for (int t = 0; t < SEQW; ++t) {
        const __bf16* hp = (t & 1) ? hB : hA;
        __bf16*       hn = (t & 1) ? hA : hB;
        lstm_step_kernel<<<dim3(HID / 16, BATCH / 16), 32, 0, stream>>>(
            X + (size_t)t * BATCH * EMB, hp, Wih_bf, Whh_bf, bih, bhh,
            C, hn, (float*)d_out, (t == SEQW - 1) ? 1 : 0);
    }
}
